// ivd_like_11879879544347
// MI455X (gfx1250) — matrix operand order repaired and checked by reading, not compiled, not run on hardware
//
#include <hip/hip_runtime.h>
#include <hip/hip_bf16.h>
#include <math.h>

#define HDIM 256
#define HID  128
#define NVOC 50000
#define NNODE 512
#define LSEQ 64
#define MAST 64
#define CCH  4
#define TTOK 8
#define KNEI 8
#define NEDGE 2048

typedef __attribute__((ext_vector_type(16))) _Float16 v16h;
typedef __attribute__((ext_vector_type(8)))  float    v8f;

// ---------------------------------------------------------------------------
// WMMA helpers (CDNA5 16x16x32 f16 -> f32)
// ---------------------------------------------------------------------------
__device__ __forceinline__ v8f wmma32(v16h a, v16h b, v8f c) {
  return __builtin_amdgcn_wmma_f32_16x16x32_f16(
      /*neg_a=*/false, a, /*neg_b=*/false, b,
      /*c_mod=*/(short)0, c, /*reuse_a=*/false, /*reuse_b=*/false);
}

// A fragment: 16 (M rows) x 32 (K), row-major f16 source (LDS).
// lane = M row (lane&15); VGPR j holds K = (j/4)*16 + (lane/16)*8 + (j%4)*2.
// Element order -> two contiguous 8-element (16B) runs: [kh..kh+7], [16+kh..].
__device__ __forceinline__ v16h afrag_f16(const _Float16* p, int rs) {
  int lane = threadIdx.x & 31;
  const _Float16* row = p + (lane & 15) * rs + ((lane >> 4) << 3);
  v16h f;
#pragma unroll
  for (int j = 0; j < 8; ++j) {
    f[j]     = row[j];
    f[j + 8] = row[16 + j];
  }
  return f;
}

__device__ __forceinline__ v16h afrag_f32(const float* p, int rs) {
  int lane = threadIdx.x & 31;
  const float* row = p + (long)(lane & 15) * rs + ((lane >> 4) << 3);
  v16h f;
#pragma unroll
  for (int j = 0; j < 8; ++j) {
    f[j]     = (_Float16)row[j];
    f[j + 8] = (_Float16)row[16 + j];
  }
  return f;
}

// B fragment: 32 (K) x 16 (N cols), B = W^T with W row-major (out x in), f16.
// lane = N col -> one W row; same K order as the A fragment: two 8-element
// (16B) runs [kh..kh+7], [16+kh..16+kh+7] with kh = (lane/16)*8.
__device__ __forceinline__ v16h bfrag_f16(const _Float16* p, int rs) {
  int lane = threadIdx.x & 31;
  const _Float16* row = p + (long)(lane & 15) * rs + ((lane >> 4) << 3);
  v16h f;
#pragma unroll
  for (int j = 0; j < 8; ++j) {
    f[j]     = row[j];
    f[j + 8] = row[16 + j];
  }
  return f;
}

__device__ __forceinline__ float sigf(float x) { return 1.0f / (1.0f + __expf(-x)); }

// one-time f32 -> f16 weight conversion
__global__ __launch_bounds__(256) void cvt16_kernel(
    const float* __restrict__ src, _Float16* __restrict__ dst, long n) {
  long i = (long)blockIdx.x * 256 + threadIdx.x;
  if (i < n) dst[i] = (_Float16)src[i];
}

// ---------------------------------------------------------------------------
// BiGRU kernel. One block = 16 batch rows, one direction (blockIdx.y).
// 8 wave32 waves; wave w owns hidden cols [16w,16w+16). Recurrent weights
// (whh slice) live in registers for the whole L-step loop; hidden state in
// LDS. Rows are independent -> no inter-block sync.
// ---------------------------------------------------------------------------
__global__ __launch_bounds__(256) void gru_kernel(
    const float* __restrict__ src, const int* __restrict__ idx,
    const _Float16* __restrict__ wih, const _Float16* __restrict__ whh,
    const float* __restrict__ bih, const float* __restrict__ bhh,
    int L, float* __restrict__ o_out) {
  int n0  = blockIdx.x * 16;
  int dir = blockIdx.y;
  wih += (size_t)dir * 3 * HID * HDIM;
  whh += (size_t)dir * 3 * HID * HID;
  bih += dir * 3 * HID;
  bhh += dir * 3 * HID;

  __shared__ _Float16 sx[16 * HDIM];
  __shared__ _Float16 sh16[16 * HID];
  __shared__ float    shf[16 * HID];

  int tid = threadIdx.x;
  int w = tid >> 5, lane = tid & 31;
  int col16 = lane & 15, rb = (lane >> 4) << 3;
  int c0 = w * 16;

  for (int e = tid; e < 16 * HID; e += 256) { shf[e] = 0.f; sh16[e] = (_Float16)0.f; }

  // recurrent weight fragments held in VGPRs across the whole scan
  v16h wb[12];
#pragma unroll
  for (int g = 0; g < 3; ++g)
#pragma unroll
    for (int kk = 0; kk < 4; ++kk)
      wb[g * 4 + kk] = bfrag_f16(whh + (long)(g * HID + c0) * HID + kk * 32, HID);

  float bir = bih[0 * HID + c0 + col16], biz = bih[1 * HID + c0 + col16],
        bin = bih[2 * HID + c0 + col16];
  float bhr = bhh[0 * HID + c0 + col16], bhz = bhh[1 * HID + c0 + col16],
        bhn = bhh[2 * HID + c0 + col16];

  for (int t = 0; t < L; ++t) {
    int pos = dir ? (L - 1 - t) : t;
    for (int e = tid; e < 16 * HDIM; e += 256) {
      int r = e >> 8, h = e & 255;
      int row = idx ? idx[(n0 + r) * L + pos] : ((n0 + r) * L + pos);
      sx[e] = (_Float16)src[(long)row * HDIM + h];
    }
    if (t + 1 < L && tid < 16) {  // prefetch next step's gathered rows
      int pos2 = dir ? (L - 2 - t) : (t + 1);
      int row2 = idx ? idx[(n0 + tid) * L + pos2] : ((n0 + tid) * L + pos2);
      __builtin_prefetch(src + (long)row2 * HDIM, 0, 1);
    }
    __syncthreads();

    v8f gi0 = {}, gi1 = {}, gi2 = {}, gh0 = {}, gh1 = {}, gh2 = {};
    for (int k = 0; k < HDIM; k += 32) {
      v16h a = afrag_f16(sx + k, HDIM);
      gi0 = wmma32(a, bfrag_f16(wih + (long)(0 * HID + c0) * HDIM + k, HDIM), gi0);
      gi1 = wmma32(a, bfrag_f16(wih + (long)(1 * HID + c0) * HDIM + k, HDIM), gi1);
      gi2 = wmma32(a, bfrag_f16(wih + (long)(2 * HID + c0) * HDIM + k, HDIM), gi2);
    }
#pragma unroll
    for (int kk = 0; kk < 4; ++kk) {
      v16h a = afrag_f16(sh16 + kk * 32, HID);
      gh0 = wmma32(a, wb[0 * 4 + kk], gh0);
      gh1 = wmma32(a, wb[1 * 4 + kk], gh1);
      gh2 = wmma32(a, wb[2 * 4 + kk], gh2);
    }
    float hnew[8];
#pragma unroll
    for (int i = 0; i < 8; ++i) {
      int rr = rb + i;
      float r = sigf(gi0[i] + bir + gh0[i] + bhr);
      float z = sigf(gi1[i] + biz + gh1[i] + bhz);
      float nn = tanhf(gi2[i] + bin + r * (gh2[i] + bhn));
      float hold = shf[rr * HID + c0 + col16];
      hnew[i] = (1.f - z) * nn + z * hold;
    }
    __syncthreads();
#pragma unroll
    for (int i = 0; i < 8; ++i) {
      int rr = rb + i;
      shf[rr * HID + c0 + col16]  = hnew[i];
      sh16[rr * HID + c0 + col16] = (_Float16)hnew[i];
      o_out[((long)(n0 + rr) * L + pos) * (2 * HID) + dir * HID + c0 + col16] = hnew[i];
    }
    __syncthreads();
  }
}

// ---------------------------------------------------------------------------
// Attention pooling: scores = softmax_l(scr[b,l,:].w + bias), out = sum_l
// wts[l] * val[b,l,:].
// ---------------------------------------------------------------------------
__global__ __launch_bounds__(256) void pool_kernel(
    const float* __restrict__ scr, long sb, long sl,
    const float* __restrict__ val, long vb, long vl,
    int L, const float* __restrict__ w, const float* __restrict__ bias,
    float* __restrict__ out) {
  int b = blockIdx.x, tid = threadIdx.x;
  __shared__ float a[64];
  if (tid < L) {
    const float* p = scr + (long)b * sb + (long)tid * sl;
    float s = 0.f;
    for (int h = 0; h < HDIM; ++h) s += p[h] * w[h];
    a[tid] = s + bias[0];
  }
  __syncthreads();
  if (tid == 0) {
    float mx = a[0];
    for (int l = 1; l < L; ++l) mx = fmaxf(mx, a[l]);
    float sum = 0.f;
    for (int l = 0; l < L; ++l) { a[l] = __expf(a[l] - mx); sum += a[l]; }
    float inv = 1.f / sum;
    for (int l = 0; l < L; ++l) a[l] *= inv;
  }
  __syncthreads();
  float s = 0.f;
  for (int l = 0; l < L; ++l) s += a[l] * val[(long)b * vb + (long)l * vl + tid];
  out[(long)b * HDIM + tid] = s;
}

// xn[n,m,:] = 0.5*label_tab[lid] + 0.5*mean_t tok_tab[tok_ids]
__global__ __launch_bounds__(256) void xn_kernel(
    const int* __restrict__ lab_ids, const int* __restrict__ tok_ids,
    const float* __restrict__ lab_tab, const float* __restrict__ tok_tab,
    float* __restrict__ xn) {
  int nm = blockIdx.x, h = threadIdx.x;
  const int* tk = tok_ids + (long)nm * TTOK;
  float s = 0.f;
#pragma unroll
  for (int t = 0; t < TTOK; ++t) s += tok_tab[(long)tk[t] * HDIM + h];
  xn[(long)nm * HDIM + h] = 0.5f * lab_tab[(long)lab_ids[nm] * HDIM + h] + 0.5f * (s * 0.125f);
}

// ---------------------------------------------------------------------------
// Child-sum TreeLSTM. One block = 16 trees; loops over the 64 node slots
// internally (children always reference earlier slots of the SAME tree).
// Wave w owns H-cols [32w,32w+32). Weights consumed as f16.
// ---------------------------------------------------------------------------
__global__ __launch_bounds__(256) void tree_kernel(
    const int* __restrict__ ch_idx, const unsigned char* __restrict__ ch_msk,
    const float* __restrict__ xn,
    const _Float16* __restrict__ wiou, const float* __restrict__ biou,
    const _Float16* __restrict__ uiou, const _Float16* __restrict__ wf,
    const float* __restrict__ bf, const _Float16* __restrict__ uf,
    float* __restrict__ h_arr, float* __restrict__ c_arr) {
  int n0 = blockIdx.x * 16;
  __shared__ _Float16 s_x[16 * HDIM];
  __shared__ _Float16 s_hs[16 * HDIM];
  __shared__ _Float16 s_ch_h[64 * HDIM];   // row = child*16 + batchrow
  __shared__ int   s_ci[64];
  __shared__ float s_cm[64];

  int tid = threadIdx.x;
  int w = tid >> 5, lane = tid & 31;
  int col16 = lane & 15, rb = (lane >> 4) << 3;

  for (int m = 0; m < MAST; ++m) {
    if (tid < 64) {
      int r = tid >> 2, k = tid & 3;
      s_ci[tid] = ch_idx[((long)(n0 + r) * MAST + m) * CCH + k];
      s_cm[tid] = ch_msk[((long)(n0 + r) * MAST + m) * CCH + k] ? 1.f : 0.f;
    }
    __syncthreads();
    for (int e = tid; e < 16 * HDIM; e += 256) {
      int r = e >> 8, h = e & 255;
      float s = 0.f;
#pragma unroll
      for (int k = 0; k < 4; ++k) {
        float mk = s_cm[r * 4 + k];
        float hv = (mk != 0.f)
            ? h_arr[((long)s_ci[r * 4 + k] * NNODE + (n0 + r)) * HDIM + h] : 0.f;
        s_ch_h[(k * 16 + r) * HDIM + h] = (_Float16)hv;
        s += hv;
      }
      s_hs[e] = (_Float16)s;
      s_x[e]  = (_Float16)xn[((long)(n0 + r) * MAST + m) * HDIM + h];
    }
    __syncthreads();

    for (int ct = 0; ct < 2; ++ct) {
      int cb = w * 32 + ct * 16;
      v8f fi = {}, fo = {}, fu = {}, fxw = {};
      v8f ff[4] = {{}, {}, {}, {}};
      for (int k = 0; k < HDIM; k += 32) {
        v16h ax = afrag_f16(s_x + k, HDIM);
        v16h ah = afrag_f16(s_hs + k, HDIM);
        fi  = wmma32(ax, bfrag_f16(wiou + (long)(0 * HDIM + cb) * HDIM + k, HDIM), fi);
        fo  = wmma32(ax, bfrag_f16(wiou + (long)(1 * HDIM + cb) * HDIM + k, HDIM), fo);
        fu  = wmma32(ax, bfrag_f16(wiou + (long)(2 * HDIM + cb) * HDIM + k, HDIM), fu);
        fi  = wmma32(ah, bfrag_f16(uiou + (long)(0 * HDIM + cb) * HDIM + k, HDIM), fi);
        fo  = wmma32(ah, bfrag_f16(uiou + (long)(1 * HDIM + cb) * HDIM + k, HDIM), fo);
        fu  = wmma32(ah, bfrag_f16(uiou + (long)(2 * HDIM + cb) * HDIM + k, HDIM), fu);
        fxw = wmma32(ax, bfrag_f16(wf + (long)cb * HDIM + k, HDIM), fxw);
#pragma unroll
        for (int c2 = 0; c2 < 4; ++c2)
          ff[c2] = wmma32(afrag_f16(s_ch_h + (long)(c2 * 16) * HDIM + k, HDIM),
                          bfrag_f16(uf + (long)cb * HDIM + k, HDIM), ff[c2]);
      }
      float bi = biou[0 * HDIM + cb + col16], bo = biou[1 * HDIM + cb + col16],
            bu = biou[2 * HDIM + cb + col16];
      float bff = bf[cb + col16];
#pragma unroll
      for (int i = 0; i < 8; ++i) {
        int rr = rb + i, n = n0 + rr, col = cb + col16;
        float ig = sigf(fi[i] + bi), og = sigf(fo[i] + bo), ug = tanhf(fu[i] + bu);
        float csum = ig * ug;
        float xw = fxw[i] + bff;
#pragma unroll
        for (int k = 0; k < 4; ++k) {
          float mk = s_cm[rr * 4 + k];
          float cc = (mk != 0.f)
              ? c_arr[((long)s_ci[rr * 4 + k] * NNODE + n) * HDIM + col] : 0.f;
          csum += sigf(xw + ff[k][i]) * cc;
        }
        c_arr[((long)m * NNODE + n) * HDIM + col] = csum;
        h_arr[((long)m * NNODE + n) * HDIM + col] = og * tanhf(csum);
      }
    }
    __syncthreads();
  }
}

// Generic WMMA GEMM: out = act(A(rows x K) @ W16(cols x K)^T + bias)
__global__ __launch_bounds__(32) void gemm_kernel(
    const float* __restrict__ A, int lda, const _Float16* __restrict__ W, int ldw,
    const float* __restrict__ bias, float* __restrict__ out, int ldo,
    int K, int dorelu) {
  int r0 = blockIdx.x * 16, c0 = blockIdx.y * 16;
  v8f acc = {};
  for (int k = 0; k < K; k += 32)
    acc = wmma32(afrag_f32(A + (long)r0 * lda + k, lda),
                 bfrag_f16(W + (long)c0 * ldw + k, ldw), acc);
  int lane = threadIdx.x & 31, col = lane & 15, rb = (lane >> 4) << 3;
  float bv = bias ? bias[c0 + col] : 0.f;
#pragma unroll
  for (int i = 0; i < 8; ++i) {
    float v = acc[i] + bv;
    if (dorelu) v = fmaxf(v, 0.f);
    out[(long)(r0 + rb + i) * ldo + c0 + col] = v;
  }
}

__global__ __launch_bounds__(256) void feats_kernel(
    const float* __restrict__ stmt, const float* __restrict__ astv,
    const float* __restrict__ varn, const float* __restrict__ vart,
    const float* __restrict__ ctxv, float* __restrict__ feats) {
  int n = blockIdx.x, h = threadIdx.x;
  feats[((long)n * 5 + 0) * HDIM + h] = stmt[(long)n * HDIM + h];
  feats[((long)n * 5 + 1) * HDIM + h] = astv[(long)n * HDIM + h];
  feats[((long)n * 5 + 2) * HDIM + h] = varn[(long)n * HDIM + h];
  feats[((long)n * 5 + 3) * HDIM + h] = vart[(long)n * HDIM + h];
  feats[((long)n * 5 + 4) * HDIM + h] = ctxv[(long)n * HDIM + h];
}

__global__ __launch_bounds__(256) void edge_kernel(
    const int* __restrict__ edges, const float* __restrict__ x,
    float* __restrict__ nei_sum, float* __restrict__ deg) {
  int e = blockIdx.x, h = threadIdx.x;
  int a = edges[2 * e], b = edges[2 * e + 1];
  atomicAdd(&nei_sum[(long)a * HDIM + h], x[(long)b * HDIM + h]);
  atomicAdd(&nei_sum[(long)b * HDIM + h], x[(long)a * HDIM + h]);
  if (h == 0) { atomicAdd(&deg[a], 1.f); atomicAdd(&deg[b], 1.f); }
}

__global__ __launch_bounds__(256) void xcat_kernel(
    const float* __restrict__ x, const float* __restrict__ neis,
    const float* __restrict__ deg, float* __restrict__ xcat) {
  int n = blockIdx.x, h = threadIdx.x;
  xcat[(long)n * 512 + h]       = x[(long)n * HDIM + h];
  xcat[(long)n * 512 + 256 + h] = neis[(long)n * HDIM + h] / fmaxf(deg[n], 1.f);
}

__global__ __launch_bounds__(256) void dis_kernel(
    const float* __restrict__ deg, float* __restrict__ dis) {
  int n = blockIdx.x * 256 + threadIdx.x;
  if (n < NNODE) dis[n] = rsqrtf(deg[n] + 1.f);  // d2 = deg + self-loop >= 1
}

// sym-normalized SPMM with self loops: out[r] += dis[r]*dis[c]*y[c]
__global__ __launch_bounds__(256) void spmm_kernel(
    const int* __restrict__ edges, const float* __restrict__ dis,
    const float* __restrict__ y, float* __restrict__ out) {
  int p = blockIdx.x, h = threadIdx.x;
  int r, c;
  if (p < NEDGE)          { r = edges[2 * p];               c = edges[2 * p + 1]; }
  else if (p < 2 * NEDGE) { r = edges[2 * (p - NEDGE) + 1]; c = edges[2 * (p - NEDGE)]; }
  else                    { r = p - 2 * NEDGE; c = r; }
  atomicAdd(&out[(long)r * HDIM + h], dis[r] * dis[c] * y[(long)c * HDIM + h]);
}

__global__ __launch_bounds__(256) void relu_kernel(float* __restrict__ x) {
  long i = (long)blockIdx.x * 256 + threadIdx.x;
  x[i] = fmaxf(x[i], 0.f);
}

__global__ __launch_bounds__(256) void gvec_kernel(
    const float* __restrict__ h2, float* __restrict__ g) {
  int h = threadIdx.x;
  float s = 0.f, mx = -1e30f;
  for (int n = 0; n < NNODE; ++n) {
    float v = h2[(long)n * HDIM + h];
    s += v; mx = fmaxf(mx, v);
  }
  g[h] = s / (float)NNODE;
  g[HDIM + h] = mx;
}

__global__ __launch_bounds__(256) void cls_kernel(
    const float* __restrict__ gvec, const float* __restrict__ w1,
    const float* __restrict__ b1, const float* __restrict__ w2,
    const float* __restrict__ b2, float* __restrict__ out) {
  int j = threadIdx.x;
  float s = 0.f;
  for (int k = 0; k < 512; ++k) s += gvec[k] * w1[(long)j * 512 + k];
  s = fmaxf(s + b1[j], 0.f);
  __shared__ float red[256];
  red[j] = s * w2[j];
  __syncthreads();
  for (int st = 128; st > 0; st >>= 1) {
    if (j < st) red[j] += red[j + st];
    __syncthreads();
  }
  if (j == 0) out[0] = red[0] + b2[0];
}

// ---------------------------------------------------------------------------
extern "C" void kernel_launch(void* const* d_in, const int* in_sizes, int n_in,
                              void* d_out, int out_size, void* d_ws, size_t ws_size,
                              hipStream_t stream) {
  (void)in_sizes; (void)n_in; (void)out_size; (void)ws_size;
  const int*   seq_tokens     = (const int*)d_in[0];
  const int*   ast_children   = (const int*)d_in[1];
  const unsigned char* ast_child_mask = (const unsigned char*)d_in[2];
  const int*   ast_label_ids  = (const int*)d_in[3];
  const int*   ast_tok_ids    = (const int*)d_in[4];
  const int*   ctx_nei        = (const int*)d_in[5];
  const int*   edges          = (const int*)d_in[6];
  const float* seq_emb        = (const float*)d_in[7];
  const float* seq_wih        = (const float*)d_in[8];
  const float* seq_whh        = (const float*)d_in[9];
  const float* seq_bih        = (const float*)d_in[10];
  const float* seq_bhh        = (const float*)d_in[11];
  const float* seq_pw         = (const float*)d_in[12];
  const float* seq_pb         = (const float*)d_in[13];
  const float* ctx_wih        = (const float*)d_in[14];
  const float* ctx_whh        = (const float*)d_in[15];
  const float* ctx_bih        = (const float*)d_in[16];
  const float* ctx_bhh        = (const float*)d_in[17];
  const float* ctx_pw         = (const float*)d_in[18];
  const float* ctx_pb         = (const float*)d_in[19];
  const float* ast_label_tab  = (const float*)d_in[20];
  const float* ast_tok_tab    = (const float*)d_in[21];
  const float* ast_pw         = (const float*)d_in[22];
  const float* ast_pb         = (const float*)d_in[23];
  const float* tl_wiou        = (const float*)d_in[24];
  const float* tl_biou        = (const float*)d_in[25];
  const float* tl_uiou        = (const float*)d_in[26];
  const float* tl_wf          = (const float*)d_in[27];
  const float* tl_bf          = (const float*)d_in[28];
  const float* tl_uf          = (const float*)d_in[29];
  const float* fa_wih         = (const float*)d_in[30];
  const float* fa_whh         = (const float*)d_in[31];
  const float* fa_bih         = (const float*)d_in[32];
  const float* fa_bhh         = (const float*)d_in[33];
  const float* fa_aw          = (const float*)d_in[34];
  const float* fa_ab          = (const float*)d_in[35];
  const float* comb_w         = (const float*)d_in[36];
  const float* comb_b         = (const float*)d_in[37];
  const float* g1_w           = (const float*)d_in[38];
  const float* g1_b           = (const float*)d_in[39];
  const float* g2_w           = (const float*)d_in[40];
  const float* g2_b           = (const float*)d_in[41];
  const float* cls_w1         = (const float*)d_in[42];
  const float* cls_b1         = (const float*)d_in[43];
  const float* cls_w2         = (const float*)d_in[44];
  const float* cls_b2         = (const float*)d_in[45];

  char* ws = (char*)d_ws;
  size_t off = 0;
  auto take = [&](size_t bytes) -> char* {
    char* p = ws + off;
    off += (bytes + 255) & ~(size_t)255;
    return p;
  };
  float* obuf  = (float*)take((size_t)NNODE * LSEQ * HDIM * 4);
  float* vecs  = (float*)take((size_t)3 * NNODE * HDIM * 4);
  float* astv  = (float*)take((size_t)NNODE * HDIM * 4);
  float* ctxv  = (float*)take((size_t)NNODE * HDIM * 4);
  float* xnb   = (float*)take((size_t)NNODE * MAST * HDIM * 4);
  float* harr  = (float*)take((size_t)MAST * NNODE * HDIM * 4);
  float* carr  = (float*)take((size_t)MAST * NNODE * HDIM * 4);
  float* feats = (float*)take((size_t)NNODE * 5 * HDIM * 4);
  float* xatt  = (float*)take((size_t)NNODE * HDIM * 4);
  float* neis  = (float*)take((size_t)NNODE * HDIM * 4);
  float* degv  = (float*)take((size_t)NNODE * 4);
  float* disv  = (float*)take((size_t)NNODE * 4);
  float* xcat  = (float*)take((size_t)NNODE * 2 * HDIM * 4);
  float* x2b   = (float*)take((size_t)NNODE * HDIM * 4);
  float* ybuf  = (float*)take((size_t)NNODE * HDIM * 4);
  float* h1b   = (float*)take((size_t)NNODE * HDIM * 4);
  float* h2b   = (float*)take((size_t)NNODE * HDIM * 4);
  float* gv    = (float*)take((size_t)2 * HDIM * 4);

  // f16 weight copies (one-time conversion per launch)
  auto cvt = [&](const float* s, long n) -> _Float16* {
    _Float16* d = (_Float16*)take((size_t)n * 2);
    cvt16_kernel<<<(int)((n + 255) / 256), 256, 0, stream>>>(s, d, n);
    return d;
  };
  _Float16* seq_wih16 = cvt(seq_wih, 3L * 2 * 3 * HID * HDIM);
  _Float16* seq_whh16 = cvt(seq_whh, 3L * 2 * 3 * HID * HID);
  _Float16* ctx_wih16 = cvt(ctx_wih, 2L * 3 * HID * HDIM);
  _Float16* ctx_whh16 = cvt(ctx_whh, 2L * 3 * HID * HID);
  _Float16* fa_wih16  = cvt(fa_wih,  2L * 3 * HID * HDIM);
  _Float16* fa_whh16  = cvt(fa_whh,  2L * 3 * HID * HID);
  _Float16* wiou16    = cvt(tl_wiou, 3L * HDIM * HDIM);
  _Float16* uiou16    = cvt(tl_uiou, 3L * HDIM * HDIM);
  _Float16* wf16      = cvt(tl_wf,   (long)HDIM * HDIM);
  _Float16* uf16      = cvt(tl_uf,   (long)HDIM * HDIM);
  _Float16* comb16    = cvt(comb_w,  (long)HDIM * 2 * HDIM);
  _Float16* g1w16     = cvt(g1_w,    (long)HDIM * HDIM);
  _Float16* g2w16     = cvt(g2_w,    (long)HDIM * HDIM);

  // --- three sequence encoders (BiGRU over tokens + attention pool) ---
  for (int enc = 0; enc < 3; ++enc) {
    gru_kernel<<<dim3(NNODE / 16, 2), 256, 0, stream>>>(
        seq_emb + (size_t)enc * NVOC * HDIM,
        seq_tokens + (size_t)enc * NNODE * LSEQ,
        seq_wih16 + (size_t)enc * 2 * 3 * HID * HDIM,
        seq_whh16 + (size_t)enc * 2 * 3 * HID * HID,
        seq_bih + (size_t)enc * 2 * 3 * HID,
        seq_bhh + (size_t)enc * 2 * 3 * HID,
        LSEQ, obuf);
    pool_kernel<<<NNODE, 256, 0, stream>>>(
        obuf, (long)LSEQ * HDIM, HDIM, obuf, (long)LSEQ * HDIM, HDIM, LSEQ,
        seq_pw + (size_t)enc * HDIM, seq_pb + enc, vecs + (size_t)enc * NNODE * HDIM);
  }

  // --- AST node features + TreeLSTM + pool ---
  xn_kernel<<<NNODE * MAST, 256, 0, stream>>>(ast_label_ids, ast_tok_ids,
                                              ast_label_tab, ast_tok_tab, xnb);
  tree_kernel<<<NNODE / 16, 256, 0, stream>>>(ast_children, ast_child_mask, xnb,
                                              wiou16, tl_biou, uiou16,
                                              wf16, tl_bf, uf16, harr, carr);
  pool_kernel<<<NNODE, 256, 0, stream>>>(
      harr, (long)HDIM, (long)NNODE * HDIM, harr, (long)HDIM, (long)NNODE * HDIM,
      MAST, ast_pw, ast_pb, astv);

  // --- context encoder over gathered stmt vectors ---
  gru_kernel<<<dim3(NNODE / 16, 2), 256, 0, stream>>>(
      vecs /* stmt_v */, ctx_nei, ctx_wih16, ctx_whh16, ctx_bih, ctx_bhh, KNEI, obuf);
  pool_kernel<<<NNODE, 256, 0, stream>>>(
      obuf, (long)KNEI * HDIM, HDIM, obuf, (long)KNEI * HDIM, HDIM, KNEI,
      ctx_pw, ctx_pb, ctxv);

  // --- feature attention BiGRU over the 5 feature vectors ---
  feats_kernel<<<NNODE, 256, 0, stream>>>(
      vecs, astv, vecs + (size_t)NNODE * HDIM, vecs + (size_t)2 * NNODE * HDIM,
      ctxv, feats);
  gru_kernel<<<dim3(NNODE / 16, 2), 256, 0, stream>>>(
      feats, nullptr, fa_wih16, fa_whh16, fa_bih, fa_bhh, 5, obuf);
  pool_kernel<<<NNODE, 256, 0, stream>>>(
      obuf, 5L * HDIM, HDIM, feats, 5L * HDIM, HDIM, 5, fa_aw, fa_ab, xatt);

  // --- neighborhood aggregation + combine MLP ---
  hipMemsetAsync(neis, 0, (size_t)NNODE * HDIM * 4, stream);
  hipMemsetAsync(degv, 0, (size_t)NNODE * 4, stream);
  edge_kernel<<<NEDGE, 256, 0, stream>>>(edges, xatt, neis, degv);
  xcat_kernel<<<NNODE, 256, 0, stream>>>(xatt, neis, degv, xcat);
  dis_kernel<<<2, 256, 0, stream>>>(degv, disv);
  gemm_kernel<<<dim3(NNODE / 16, HDIM / 16), 32, 0, stream>>>(
      xcat, 2 * HDIM, comb16, 2 * HDIM, comb_b, x2b, HDIM, 2 * HDIM, 1);

  // --- GCN layer 1 ---
  gemm_kernel<<<dim3(NNODE / 16, HDIM / 16), 32, 0, stream>>>(
      x2b, HDIM, g1w16, HDIM, g1_b, ybuf, HDIM, HDIM, 0);
  hipMemsetAsync(h1b, 0, (size_t)NNODE * HDIM * 4, stream);
  spmm_kernel<<<2 * NEDGE + NNODE, 256, 0, stream>>>(edges, disv, ybuf, h1b);
  relu_kernel<<<NNODE, 256, 0, stream>>>(h1b);

  // --- GCN layer 2 ---
  gemm_kernel<<<dim3(NNODE / 16, HDIM / 16), 32, 0, stream>>>(
      h1b, HDIM, g2w16, HDIM, g2_b, ybuf, HDIM, HDIM, 0);
  hipMemsetAsync(h2b, 0, (size_t)NNODE * HDIM * 4, stream);
  spmm_kernel<<<2 * NEDGE + NNODE, 256, 0, stream>>>(edges, disv, ybuf, h2b);
  relu_kernel<<<NNODE, 256, 0, stream>>>(h2b);

  // --- graph readout + classifier ---
  gvec_kernel<<<1, 256, 0, stream>>>(h2b, gv);
  cls_kernel<<<1, 256, 0, stream>>>(gv, cls_w1, cls_b1, cls_w2, cls_b2,
                                    (float*)d_out);
}
